// LocationAwareAttention_19834158973127
// MI455X (gfx1250) — matrix operand order repaired and checked by reading, compiled, not run on hardware
//
#include <hip/hip_runtime.h>
#include <hip/hip_bf16.h>

// ---------------- problem constants ----------------
#define B    64
#define T    2000
#define D    512      // Denc == Ddec == A
#define FNUM 100
#define KW   101      // conv kernel width
#define KPAD 128      // conv band K padded to multiple of 32
#define KTOT (D + KPAD)   // 640 stacked K
#define NKT  (KTOT / 32)  // 20 K-steps of 32
#define TPAD 2176     // 64 leading zeros + 2000 + trailing pad
#define RSTRIDE 648   // LDS A row stride in bf16 (640 + 8 pad, 16B-aligned rows)

#define SLAB_ELEMS (32 * 512)          // one K-slab of B: 32 k x 512 n bf16 = 32 KB
#define A_LDS_BYTES (64 * RSTRIDE * 2) // 82944
#define B_LDS_BYTES (2 * SLAB_ELEMS * 2)
#define SMEM_BYTES  (A_LDS_BYTES + B_LDS_BYTES + 64 * 4)  // 148736 < 320 KB WGP LDS

typedef __attribute__((ext_vector_type(16))) __bf16 v16bf;
typedef __attribute__((ext_vector_type(8)))  float  v8f;
typedef __attribute__((__vector_size__(4 * sizeof(int)))) int i32x4;

union FragA {                       // A fragment: two 16B runs (k, k+8)
    struct { uint4 lo, hi; } u;
    v16bf v;
};
union FragB {                       // B fragment: four 8B units (conflict-free ds_load_b64)
    struct { uint2 q0, q1, q2, q3; } u;
    v16bf v;
};

#if defined(__gfx1250__) && __has_builtin(__builtin_amdgcn_global_load_async_to_lds_b128) && \
    __has_builtin(__builtin_amdgcn_s_wait_asynccnt)
#define USE_ASYNC 1
#else
#define USE_ASYNC 0
#endif

// copy one 32 KB B-slab global -> LDS (async DMA if available, VGPR relay otherwise)
__device__ __forceinline__ void slab_copy(const __bf16* __restrict__ g, __bf16* l, int tid) {
#if USE_ASYNC
    const __attribute__((address_space(1))) char* gp =
        (const __attribute__((address_space(1))) char*)g;
    __attribute__((address_space(3))) char* lp =
        (__attribute__((address_space(3))) char*)l;
    for (int i = tid; i < 2048; i += 256)
        __builtin_amdgcn_global_load_async_to_lds_b128(
            (__attribute__((address_space(1))) i32x4*)(gp + i * 16),
            (__attribute__((address_space(3))) i32x4*)(lp + i * 16), 0, 0);
#else
    for (int i = tid; i < 2048; i += 256)
        *(uint4*)((char*)l + i * 16) = *(const uint4*)((const char*)g + i * 16);
#endif
}

__device__ __forceinline__ void async_wait_all() {
#if USE_ASYNC
    __builtin_amdgcn_s_wait_asynccnt(0);
#endif
}

// ---------------- prep 1: pack stacked B matrix into WMMA B-fragment layout ----------------
// Fragment element i (i=0..15) of lane L in tile (kt,nt) holds
//   Wstack[kt*32 + (L>>4)*16 + i][nt*16 + (L&15)]
// stored as 8-byte units: wsB[(((kt*32+nt)*4 + i>>2)*32 + L)*4 + (i&3)]
// Wstack rows 0..511   = W_enc[k][a]
// Wstack rows 512..639 = Wc[kk][a] = sum_f conv_w[f][kk] * W_att[f][a]   (kk<101, else 0)
__global__ __launch_bounds__(256) void pack_b_kernel(const float* __restrict__ W_enc,
                                                     const float* __restrict__ conv_w,
                                                     const float* __restrict__ W_att,
                                                     __bf16* __restrict__ wsB) {
    int tile = blockIdx.x * 8 + (threadIdx.x >> 5);   // 640 tiles = 20 kt * 32 nt
    int lane = threadIdx.x & 31;
    if (tile >= NKT * 32) return;
    int kt  = tile >> 5;
    int nt  = tile & 31;
    int col = nt * 16 + (lane & 15);
    int kbase = kt * 32 + (lane >> 4) * 16;
#pragma unroll 4
    for (int i = 0; i < 16; ++i) {
        int k = kbase + i;
        float v = 0.0f;
        if (k < D) {
            v = W_enc[(size_t)k * D + col];
        } else {
            int kk = k - D;
            if (kk < KW) {
                for (int f = 0; f < FNUM; ++f)
                    v += conv_w[f * KW + kk] * W_att[f * D + col];
            }
        }
        int q = i >> 2, e = i & 3;
        wsB[(((size_t)tile * 4 + q) * 32 + lane) * 4 + e] = (__bf16)v;
    }
}

// ---------------- prep 2: pd[b][a] = input_dec[b] . W_dec[:,a] + b_att[a] ----------------
__global__ __launch_bounds__(256) void pd_kernel(const float* __restrict__ input_dec,
                                                 const float* __restrict__ W_dec,
                                                 const float* __restrict__ b_att,
                                                 float* __restrict__ pd) {
    int b = blockIdx.x;
    for (int a = threadIdx.x; a < D; a += 256) {
        float acc = b_att[a];
        const float* xd = input_dec + (size_t)b * D;
        for (int d = 0; d < D; ++d)
            acc += xd[d] * W_dec[(size_t)d * D + a];
        pd[(size_t)b * D + a] = acc;
    }
}

// ---------------- prep 3: zero-padded bf16 prev_att ----------------
__global__ __launch_bounds__(256) void ppad_kernel(const float* __restrict__ prev_att,
                                                   __bf16* __restrict__ ppad) {
    int b = blockIdx.x;
    int i = blockIdx.y * 256 + threadIdx.x;
    if (i >= TPAD) return;
    float v = 0.0f;
    if (i >= 64 && i < 64 + T) v = prev_att[(size_t)b * T + (i - 64)];
    ppad[(size_t)b * TPAD + i] = (__bf16)v;
}

// ---------------- main: score[b][t] via stacked-K bf16 WMMA GEMM + tanh epilogue ----------------
// grid (B, 32 t-tiles of 64 rows), block 256 (8 waves as 2(M) x 4(N)).
// B K-slabs double-buffered in LDS via async global->LDS DMA, shared by all 8 waves.
__global__ __launch_bounds__(256) void score_kernel(const float* __restrict__ enc,
                                                    const __bf16* __restrict__ ppad,
                                                    const __bf16* __restrict__ wsB,
                                                    const float* __restrict__ pd,
                                                    const float* __restrict__ W_out,
                                                    const float* __restrict__ b_out,
                                                    float* __restrict__ score_out) {
    extern __shared__ char smem[];
    __bf16* A_lds     = (__bf16*)smem;                           // [64][RSTRIDE]
    __bf16* B_lds     = (__bf16*)(smem + A_LDS_BYTES);           // 2 x SLAB_ELEMS
    float*  score_lds = (float*)(smem + A_LDS_BYTES + B_LDS_BYTES);

    const int tid   = threadIdx.x;
    const int b     = blockIdx.x;
    const int tbase = blockIdx.y * 64;

    // kick off async DMA of the first B slab while we stage + convert A
    slab_copy(wsB, B_lds, tid);

    // ---- stage A tile (enc cols 0..511 + conv band cols 512..639) as bf16 into LDS ----
    const float* encb = enc + (size_t)b * T * D;
    for (int i = tid; i < 64 * 128; i += 256) {
        int row = i >> 7;
        int c4  = (i & 127) * 4;
        int tg  = min(tbase + row, T - 1);
        const float4 g = *(const float4*)(encb + (size_t)tg * D + c4);
        __bf16* dst = A_lds + row * RSTRIDE + c4;
        dst[0] = (__bf16)g.x; dst[1] = (__bf16)g.y;
        dst[2] = (__bf16)g.z; dst[3] = (__bf16)g.w;
    }
    const __bf16* pb = ppad + (size_t)b * TPAD;
    for (int i = tid; i < 64 * 128; i += 256) {
        int row = i >> 7;
        int kk  = i & 127;
        int tg  = min(tbase + row, T - 1);
        A_lds[row * RSTRIDE + D + kk] = pb[tg + 14 + kk];   // prev[t-50+kk] with 64-lead pad
    }
    if (tid < 64) score_lds[tid] = 0.0f;
    async_wait_all();
    __syncthreads();

    // ---- WMMA main loop (double-buffered B slabs) ----
    const int wave = tid >> 5, lane = tid & 31;
    const int wm = wave >> 2, wn = wave & 3;       // 2 x 4 wave grid
    const int halfl = lane >> 4, r15 = lane & 15;

    v8f acc[2][8];
#pragma unroll
    for (int ml = 0; ml < 2; ++ml)
#pragma unroll
        for (int nt = 0; nt < 8; ++nt)
            acc[ml][nt] = (v8f){0.f, 0.f, 0.f, 0.f, 0.f, 0.f, 0.f, 0.f};

    int buf = 0;
    for (int kt = 0; kt < NKT; ++kt) {
        // prefetch next slab into the other buffer (overlaps WMMA below)
        if (kt + 1 < NKT)
            slab_copy(wsB + (size_t)(kt + 1) * SLAB_ELEMS,
                      B_lds + (buf ^ 1) * SLAB_ELEMS, tid);

        v16bf afrag[2];
#pragma unroll
        for (int ml = 0; ml < 2; ++ml) {
            int row = wm * 32 + ml * 16 + r15;
            const __bf16* ap = A_lds + row * RSTRIDE + kt * 32 + halfl * 16;
            FragA fa;
            fa.u.lo = *(const uint4*)ap;          // k = kb .. kb+7
            fa.u.hi = *(const uint4*)(ap + 8);    // k = kb+8 .. kb+15
            afrag[ml] = fa.v;
        }

        const __bf16* bslab = B_lds + buf * SLAB_ELEMS;
#pragma unroll
        for (int nt = 0; nt < 8; ++nt) {
            // four ds_load_b64, 8B lane stride -> 64 distinct banks, conflict-free
            const __bf16* bs = bslab + ((size_t)(wn * 8 + nt) * 4 * 32 + lane) * 4;
            FragB fb;
            fb.u.q0 = *(const uint2*)(bs);
            fb.u.q1 = *(const uint2*)(bs + 128);
            fb.u.q2 = *(const uint2*)(bs + 256);
            fb.u.q3 = *(const uint2*)(bs + 384);
            v16bf bfrag = fb.v;
            acc[0][nt] = __builtin_amdgcn_wmma_f32_16x16x32_bf16(
                false, afrag[0], false, bfrag, (short)0, acc[0][nt], false, false);
            acc[1][nt] = __builtin_amdgcn_wmma_f32_16x16x32_bf16(
                false, afrag[1], false, bfrag, (short)0, acc[1][nt], false, false);
        }

        async_wait_all();    // next slab landed
        __syncthreads();     // all waves done reading current slab + see new one
        buf ^= 1;
    }

    // ---- epilogue: score_row = sum_a tanh(acc + pd[a]) * W_out[a] ----
    const float* pdb = pd + (size_t)b * D;
#pragma unroll
    for (int ml = 0; ml < 2; ++ml) {
        float rsum[8] = {0.f, 0.f, 0.f, 0.f, 0.f, 0.f, 0.f, 0.f};
#pragma unroll
        for (int nt = 0; nt < 8; ++nt) {
            int col = wn * 128 + nt * 16 + r15;
            float pdv = pdb[col];
            float wo  = W_out[col];
#pragma unroll
            for (int j = 0; j < 8; ++j)
                rsum[j] += tanhf(acc[ml][nt][j] + pdv) * wo;
        }
        int rowb = wm * 32 + ml * 16 + halfl * 8;
#pragma unroll
        for (int j = 0; j < 8; ++j) {
            float v = rsum[j];
            v += __shfl_xor(v, 1, 32);
            v += __shfl_xor(v, 2, 32);
            v += __shfl_xor(v, 4, 32);
            v += __shfl_xor(v, 8, 32);
            if (r15 == 0) atomicAdd(&score_lds[rowb + j], v);
        }
    }
    __syncthreads();

    if (tid < 64) {
        int t = tbase + tid;
        if (t < T) score_out[(size_t)b * T + t] = score_lds[tid] + b_out[0];
    }
}

// ---------------- softmax (length-masked) + context ----------------
// grid (B, 2 d-halves), block 256
__global__ __launch_bounds__(256) void softmax_ctx_kernel(const float* __restrict__ score,
                                                          const int* __restrict__ lens,
                                                          const float* __restrict__ enc,
                                                          float* __restrict__ ctx_out,
                                                          float* __restrict__ att_out) {
    __shared__ float w[2048];
    __shared__ float red[256];
    const int b = blockIdx.x, dh = blockIdx.y, tid = threadIdx.x;
    const int len = lens[b];

    float lmax = -3.4e38f;
    for (int t = tid; t < T; t += 256) {
        float s = (t < len) ? score[(size_t)b * T + t] : -3.4e38f;
        w[t] = s;
        lmax = fmaxf(lmax, s);
    }
    red[tid] = lmax;
    __syncthreads();
    for (int s = 128; s > 0; s >>= 1) {
        if (tid < s) red[tid] = fmaxf(red[tid], red[tid + s]);
        __syncthreads();
    }
    float m = red[0];
    __syncthreads();

    float lsum = 0.0f;
    for (int t = tid; t < T; t += 256) {
        float e = (t < len) ? __expf(w[t] - m) : 0.0f;
        w[t] = e;
        lsum += e;
    }
    red[tid] = lsum;
    __syncthreads();
    for (int s = 128; s > 0; s >>= 1) {
        if (tid < s) red[tid] += red[tid + s];
        __syncthreads();
    }
    float inv = 1.0f / red[0];
    __syncthreads();

    for (int t = tid; t < T; t += 256) {
        float wt = w[t] * inv;
        w[t] = wt;
        if (dh == 0) att_out[(size_t)b * T + t] = wt;
    }
    __syncthreads();

    // context: each thread owns one d column; coalesced 2 KB row reads
    const int d = dh * 256 + tid;
    const float* encb = enc + (size_t)b * T * D + d;
    float c = 0.0f;
#pragma unroll 4
    for (int t = 0; t < len; ++t)
        c += w[t] * encb[(size_t)t * D];
    ctx_out[(size_t)b * D + d] = c;
}

// ---------------- launcher ----------------
extern "C" void kernel_launch(void* const* d_in, const int* in_sizes, int n_in,
                              void* d_out, int out_size, void* d_ws, size_t ws_size,
                              hipStream_t stream) {
    const float* input_enc   = (const float*)d_in[0];
    const int*   enc_lengths = (const int*)d_in[1];
    const float* input_dec   = (const float*)d_in[2];
    const float* prev_att    = (const float*)d_in[3];
    const float* W_enc       = (const float*)d_in[4];
    const float* W_dec       = (const float*)d_in[5];
    const float* conv_w      = (const float*)d_in[6];
    const float* W_att       = (const float*)d_in[7];
    const float* b_att       = (const float*)d_in[8];
    const float* W_out       = (const float*)d_in[9];
    const float* b_out       = (const float*)d_in[10];

    float* ctx_out = (float*)d_out;                // [B, D]
    float* att_out = ctx_out + (size_t)B * D;      // [B, T]

    // workspace layout
    char*   ws    = (char*)d_ws;
    float*  pd    = (float*)ws;                                   // B*D*4       = 131072
    __bf16* wsB   = (__bf16*)(ws + 131072);                       // 20*32*512*2 = 655360
    __bf16* ppad  = (__bf16*)(ws + 131072 + 655360);              // B*TPAD*2    = 278528
    float*  score = (float*)(ws + 131072 + 655360 + 278528);      // B*T*4       = 512000

    pack_b_kernel<<<dim3((NKT * 32) / 8), 256, 0, stream>>>(W_enc, conv_w, W_att, wsB);
    pd_kernel<<<dim3(B), 256, 0, stream>>>(input_dec, W_dec, b_att, pd);
    ppad_kernel<<<dim3(B, (TPAD + 255) / 256), 256, 0, stream>>>(prev_att, ppad);

    score_kernel<<<dim3(B, (T + 63) / 64), 256, SMEM_BYTES, stream>>>(
        input_enc, ppad, wsB, pd, W_out, b_out, score);

    softmax_ctx_kernel<<<dim3(B, 2), 256, 0, stream>>>(
        score, enc_lengths, input_enc, ctx_out, att_out);
}
